// BidirectionalAttention_12077448036716
// MI455X (gfx1250) — matrix operand order repaired and checked by reading, not compiled, not run on hardware
//
#include <hip/hip_runtime.h>
#include <hip/hip_bf16.h>
#include <stdint.h>

#define NB 8
#define LL 2048
#define RR 2048
#define DD 1024

#define BM 128
#define BN 128
#define BK 32
#define LDT 40   // LDS row stride in bf16 elements (32 + 8 pad; 80B keeps 16B alignment)

#ifndef __has_builtin
#define __has_builtin(x) 0
#endif
#if __has_builtin(__builtin_amdgcn_global_load_async_to_lds_b128) && \
    __has_builtin(__builtin_amdgcn_s_wait_asynccnt)
#define HAVE_ASYNC_LDS 1
#else
#define HAVE_ASYNC_LDS 0
#endif

typedef __bf16 v16bf __attribute__((ext_vector_type(16)));
typedef __bf16 v8bf  __attribute__((ext_vector_type(8)));
typedef __bf16 v4bf  __attribute__((ext_vector_type(4)));
typedef float  v8f   __attribute__((ext_vector_type(8)));
typedef float  v4f   __attribute__((ext_vector_type(4)));

#if HAVE_ASYNC_LDS
typedef int v4i_vs __attribute__((vector_size(4 * sizeof(int))));
typedef __attribute__((address_space(1))) v4i_vs* g4_ptr_t;   // global (addrspace 1)
typedef __attribute__((address_space(3))) v4i_vs* l4_ptr_t;   // LDS    (addrspace 3)
#define TO_GLOBAL(p) ((g4_ptr_t)(uintptr_t)(p))
#define TO_LDS(p)    ((l4_ptr_t)(uint32_t)(uintptr_t)(p))
#endif

static __device__ __forceinline__ v16bf frag_cat(v8bf a, v8bf b) {
  return __builtin_shufflevector(a, b, 0,1,2,3,4,5,6,7,8,9,10,11,12,13,14,15);
}

// A fragment (16x32 bf16) from LDS tile s (row-major, stride LDT), rows [row0,row0+16).
// ISA layout: lanes 0-15 hold row M=lane, K {0..7,16..23}; lanes 16-31 hold K {8..15,24..31}.
static __device__ __forceinline__ v16bf load_frag_a(const __bf16* s, int row0, int lane) {
  const __bf16* p = s + (row0 + (lane & 15)) * LDT + ((lane >> 4) << 3);
  return frag_cat(*(const v8bf*)p, *(const v8bf*)(p + 16));
}

// B fragment (32x16 bf16); LDS tile holds B^T (column-of-B per row): s[n*LDT + k].
// ISA layout: lanes 0-15 hold column N=lane, K {0..7,16..23}; lanes 16-31 hold K {8..15,24..31}.
static __device__ __forceinline__ v16bf load_frag_b(const __bf16* s, int col0, int lane) {
  const __bf16* p = s + (col0 + (lane & 15)) * LDT + ((lane >> 4) << 3);
  return frag_cat(*(const v8bf*)p, *(const v8bf*)(p + 16));
}

// ---------------------------------------------------------------------------
// K0: scale by w and convert to bf16 (both scaled and plain copies)
// ---------------------------------------------------------------------------
__global__ __launch_bounds__(256) void bidir_scale_convert_kernel(
    const float* __restrict__ lhs, const float* __restrict__ rhs,
    const float* __restrict__ w_lhs, const float* __restrict__ w_rhs,
    __bf16* __restrict__ lhsw, __bf16* __restrict__ rhsw,
    __bf16* __restrict__ lhsb, __bf16* __restrict__ rhsb)
{
  size_t idx = (size_t)blockIdx.x * 256 + threadIdx.x;   // group of 4 floats
  const size_t total = (size_t)NB * LL * DD / 4;
  if (idx >= total) return;
  int dbase = (int)((idx * 4) & (DD - 1));
  v4f wl = *(const v4f*)(w_lhs + dbase);
  v4f wr = *(const v4f*)(w_rhs + dbase);
  v4f lv = ((const v4f*)lhs)[idx];
  v4f rv = ((const v4f*)rhs)[idx];
  v4bf a, b, c, d;
#pragma unroll
  for (int u = 0; u < 4; ++u) {
    a[u] = (__bf16)(lv[u] * wl[u]);
    b[u] = (__bf16)(rv[u] * wr[u]);
    c[u] = (__bf16)lv[u];
    d[u] = (__bf16)rv[u];
  }
  ((v4bf*)lhsw)[idx] = a;
  ((v4bf*)rhsw)[idx] = b;
  ((v4bf*)lhsb)[idx] = c;
  ((v4bf*)rhsb)[idx] = d;
}

// ---------------------------------------------------------------------------
// K1: scores S[n,l,r] = lhsw[n,l,:] . rhsw[n,r,:]   (bf16 WMMA, f32 acc)
// Async global->LDS double-buffered pipeline when the gfx1250 builtins exist.
// ---------------------------------------------------------------------------
__global__ __launch_bounds__(256) void bidir_scores_kernel(
    const __bf16* __restrict__ A, const __bf16* __restrict__ B, float* __restrict__ S)
{
  const int n  = blockIdx.z;
  const int m0 = blockIdx.y * BM;
  const int n0 = blockIdx.x * BN;
  const int tid  = threadIdx.x;
  const int lane = tid & 31;
  const int wave = tid >> 5;
  const int wm = (wave >> 2) * 64;
  const int wn = (wave & 3) * 32;

  const __bf16* ga = A + ((size_t)n * LL + m0) * DD;
  const __bf16* gb = B + ((size_t)n * RR + n0) * DD;

  const int lr = tid >> 1;
  const int lc = (tid & 1) * 16;

  v8f acc[4][2];
#pragma unroll
  for (int i = 0; i < 4; ++i)
#pragma unroll
    for (int j = 0; j < 2; ++j)
#pragma unroll
      for (int v = 0; v < 8; ++v) acc[i][j][v] = 0.0f;

#if HAVE_ASYNC_LDS
  __shared__ __bf16 sA[2][BM * LDT];
  __shared__ __bf16 sB[2][BN * LDT];

  auto issue = [&](int buf, int k0) {
    const __bf16* pa = ga + (size_t)lr * DD + k0 + lc;
    const __bf16* pb = gb + (size_t)lr * DD + k0 + lc;
    __bf16* la = &sA[buf][lr * LDT + lc];
    __bf16* lb = &sB[buf][lr * LDT + lc];
    __builtin_amdgcn_global_load_async_to_lds_b128(TO_GLOBAL(pa),     TO_LDS(la),     0, 0);
    __builtin_amdgcn_global_load_async_to_lds_b128(TO_GLOBAL(pa + 8), TO_LDS(la + 8), 0, 0);
    __builtin_amdgcn_global_load_async_to_lds_b128(TO_GLOBAL(pb),     TO_LDS(lb),     0, 0);
    __builtin_amdgcn_global_load_async_to_lds_b128(TO_GLOBAL(pb + 8), TO_LDS(lb + 8), 0, 0);
  };

  constexpr int NIT = DD / BK;
  issue(0, 0);
  for (int i = 0; i < NIT; ++i) {
    const int cur = i & 1;
    if (i + 1 < NIT) {
      issue(cur ^ 1, (i + 1) * BK);            // prefill next buffer (async)
      __builtin_amdgcn_s_wait_asynccnt(4);     // stage i complete (4 newer in flight)
    } else {
      __builtin_amdgcn_s_wait_asynccnt(0);
    }
    __syncthreads();

    const __bf16* cA = sA[cur];
    const __bf16* cB = sB[cur];
    v16bf fa[4], fb[2];
#pragma unroll
    for (int i2 = 0; i2 < 4; ++i2) fa[i2] = load_frag_a(cA, wm + i2 * 16, lane);
#pragma unroll
    for (int j2 = 0; j2 < 2; ++j2) fb[j2] = load_frag_b(cB, wn + j2 * 16, lane);
#pragma unroll
    for (int i2 = 0; i2 < 4; ++i2)
#pragma unroll
      for (int j2 = 0; j2 < 2; ++j2)
        acc[i2][j2] = __builtin_amdgcn_wmma_f32_16x16x32_bf16(
            false, fa[i2], false, fb[j2], (short)0, acc[i2][j2], false, false);
    __syncthreads();                            // stage i fully consumed before reuse
  }
#else
  __shared__ __bf16 sA[BM * LDT];
  __shared__ __bf16 sB[BN * LDT];

  for (int k0 = 0; k0 < DD; k0 += BK) {
    const __bf16* pa = ga + (size_t)lr * DD + k0 + lc;
    const __bf16* pb = gb + (size_t)lr * DD + k0 + lc;
    if (k0 + BK < DD) {
      __builtin_prefetch(pa + BK, 0, 1);
      __builtin_prefetch(pb + BK, 0, 1);
    }
    *(v8bf*)(sA + lr * LDT + lc)     = *(const v8bf*)pa;
    *(v8bf*)(sA + lr * LDT + lc + 8) = *(const v8bf*)(pa + 8);
    *(v8bf*)(sB + lr * LDT + lc)     = *(const v8bf*)pb;
    *(v8bf*)(sB + lr * LDT + lc + 8) = *(const v8bf*)(pb + 8);
    __syncthreads();

    v16bf fa[4], fb[2];
#pragma unroll
    for (int i = 0; i < 4; ++i) fa[i] = load_frag_a(sA, wm + i * 16, lane);
#pragma unroll
    for (int j = 0; j < 2; ++j) fb[j] = load_frag_b(sB, wn + j * 16, lane);
#pragma unroll
    for (int i = 0; i < 4; ++i)
#pragma unroll
      for (int j = 0; j < 2; ++j)
        acc[i][j] = __builtin_amdgcn_wmma_f32_16x16x32_bf16(
            false, fa[i], false, fb[j], (short)0, acc[i][j], false, false);
    __syncthreads();
  }
#endif

  float* out = S + ((size_t)n * LL + m0) * RR + n0;
#pragma unroll
  for (int i = 0; i < 4; ++i)
#pragma unroll
    for (int j = 0; j < 2; ++j) {
      const int r0 = wm + i * 16 + ((lane >> 4) << 3);
      const int c  = wn + j * 16 + (lane & 15);
#pragma unroll
      for (int v = 0; v < 8; ++v)
        out[(size_t)(r0 + v) * RR + c] = acc[i][j][v];
    }
}

// ---------------------------------------------------------------------------
// K2a: per-row (over R) online softmax stats -> max, 1/sum
// ---------------------------------------------------------------------------
__global__ __launch_bounds__(256) void bidir_row_stats_kernel(
    const float* __restrict__ S, float* __restrict__ rmax, float* __restrict__ rinv)
{
  const int row = blockIdx.x;                 // n*LL + l
  const float* p = S + (size_t)row * RR;
  float m = -__builtin_inff(), s = 0.0f;
  for (int c = threadIdx.x; c < RR; c += 256) {
    float x = p[c];
    float mn = fmaxf(m, x);
    s = s * __expf(m - mn) + __expf(x - mn);
    m = mn;
  }
  __shared__ float sm[256];
  __shared__ float ss[256];
  sm[threadIdx.x] = m; ss[threadIdx.x] = s;
  __syncthreads();
  for (int st = 128; st > 0; st >>= 1) {
    if ((int)threadIdx.x < st) {
      float m2 = sm[threadIdx.x + st], s2 = ss[threadIdx.x + st];
      float mo = sm[threadIdx.x],      so = ss[threadIdx.x];
      float mn = fmaxf(mo, m2);
      sm[threadIdx.x] = mn;
      ss[threadIdx.x] = so * __expf(mo - mn) + s2 * __expf(m2 - mn);
    }
    __syncthreads();
  }
  if (threadIdx.x == 0) { rmax[row] = sm[0]; rinv[row] = 1.0f / ss[0]; }
}

// ---------------------------------------------------------------------------
// K2b: per-column (over L) online softmax stats -> max, 1/sum
// ---------------------------------------------------------------------------
__global__ __launch_bounds__(256) void bidir_col_stats_kernel(
    const float* __restrict__ S, float* __restrict__ cmax, float* __restrict__ cinv)
{
  const int n = blockIdx.y;
  const int c = blockIdx.x * 256 + threadIdx.x;
  const float* p = S + (size_t)n * LL * RR + c;
  float m = -__builtin_inff(), s = 0.0f;
  for (int l = 0; l < LL; ++l) {
    float x = p[(size_t)l * RR];
    float mn = fmaxf(m, x);
    s = s * __expf(m - mn) + __expf(x - mn);
    m = mn;
  }
  cmax[n * RR + c] = m;
  cinv[n * RR + c] = 1.0f / s;
}

// ---------------------------------------------------------------------------
// K3: lhs_ctx[l,d] = sum_r P_row[l,r] * rhs[r,d]; P normalized on the fly.
// Writes directly into second half of out_lhs.
// ---------------------------------------------------------------------------
__global__ __launch_bounds__(256) void bidir_lhs_ctx_kernel(
    const float* __restrict__ S, const float* __restrict__ rmax, const float* __restrict__ rinv,
    const __bf16* __restrict__ V, float* __restrict__ out)
{
  __shared__ __bf16 sA[BM * LDT];
  __shared__ __bf16 sB[BN * LDT];

  const int n  = blockIdx.z;
  const int m0 = blockIdx.y * BM;   // l
  const int n0 = blockIdx.x * BN;   // d
  const int tid  = threadIdx.x;
  const int lane = tid & 31;
  const int wave = tid >> 5;
  const int wm = (wave >> 2) * 64;
  const int wn = (wave & 3) * 32;

  const int lr = tid >> 1;              // l row in tile
  const int lc = (tid & 1) * 16;        // r col base
  const float am = rmax[n * LL + m0 + lr];
  const float ai = rinv[n * LL + m0 + lr];
  const float* arow = S + ((size_t)(n * LL + m0 + lr)) * RR;

  const int bk = tid >> 3;              // 0..31 (r within k-tile)
  const int bc = (tid & 7) * 16;        // d col base
  const __bf16* gB = V + (size_t)n * RR * DD + n0 + bc;

  v8f acc[4][2];
#pragma unroll
  for (int i = 0; i < 4; ++i)
#pragma unroll
    for (int j = 0; j < 2; ++j)
#pragma unroll
      for (int v = 0; v < 8; ++v) acc[i][j][v] = 0.0f;

  for (int k0 = 0; k0 < RR; k0 += BK) {
    if (k0 + BK < RR) {
      __builtin_prefetch(arow + k0 + BK + lc, 0, 1);
      __builtin_prefetch(gB + (size_t)(k0 + BK + bk) * DD, 0, 1);
    }
    // A: normalized probability tile 128(l) x 32(r) -> bf16
#pragma unroll
    for (int u = 0; u < 16; u += 4) {
      v4f x = *(const v4f*)(arow + k0 + lc + u);
      v4bf o;
#pragma unroll
      for (int q = 0; q < 4; ++q) o[q] = (__bf16)(__expf(x[q] - am) * ai);
      *(v4bf*)(sA + lr * LDT + lc + u) = o;
    }
    // B: V tile 32(r) x 128(d), stored transposed: sB[d][r]
    {
      const __bf16* pv = gB + (size_t)(k0 + bk) * DD;
      v8bf b0 = *(const v8bf*)pv;
      v8bf b1 = *(const v8bf*)(pv + 8);
#pragma unroll
      for (int u = 0; u < 8; ++u) sB[(bc + u) * LDT + bk] = b0[u];
#pragma unroll
      for (int u = 0; u < 8; ++u) sB[(bc + 8 + u) * LDT + bk] = b1[u];
    }
    __syncthreads();

    v16bf fa[4], fb[2];
#pragma unroll
    for (int i = 0; i < 4; ++i) fa[i] = load_frag_a(sA, wm + i * 16, lane);
#pragma unroll
    for (int j = 0; j < 2; ++j) fb[j] = load_frag_b(sB, wn + j * 16, lane);
#pragma unroll
    for (int i = 0; i < 4; ++i)
#pragma unroll
      for (int j = 0; j < 2; ++j)
        acc[i][j] = __builtin_amdgcn_wmma_f32_16x16x32_bf16(
            false, fa[i], false, fb[j], (short)0, acc[i][j], false, false);
    __syncthreads();
  }

  float* o = out + ((size_t)(n * LL + m0)) * (2 * DD) + DD + n0;
#pragma unroll
  for (int i = 0; i < 4; ++i)
#pragma unroll
    for (int j = 0; j < 2; ++j) {
      const int r0 = wm + i * 16 + ((lane >> 4) << 3);
      const int c  = wn + j * 16 + (lane & 15);
#pragma unroll
      for (int v = 0; v < 8; ++v)
        o[(size_t)(r0 + v) * (2 * DD) + c] = acc[i][j][v];
    }
}

// ---------------------------------------------------------------------------
// K4: rhs_ctx[r,d] = sum_l P_col[l,r] * lhs[l,d]; A = P_col^T built by
// transpose-on-store of coalesced S reads. Writes second half of out_rhs.
// ---------------------------------------------------------------------------
__global__ __launch_bounds__(256) void bidir_rhs_ctx_kernel(
    const float* __restrict__ S, const float* __restrict__ cmax, const float* __restrict__ cinv,
    const __bf16* __restrict__ V, float* __restrict__ out)
{
  __shared__ __bf16 sA[BM * LDT];
  __shared__ __bf16 sB[BN * LDT];

  const int n  = blockIdx.z;
  const int m0 = blockIdx.y * BM;   // r
  const int n0 = blockIdx.x * BN;   // d
  const int tid  = threadIdx.x;
  const int lane = tid & 31;
  const int wave = tid >> 5;
  const int wm = (wave >> 2) * 64;
  const int wn = (wave & 3) * 32;

  const int al = tid >> 3;              // 0..31 (l within k-tile)
  const int ac = (tid & 7) * 16;        // r col base within tile
  const float* cmp = cmax + n * RR + m0 + ac;
  const float* cip = cinv + n * RR + m0 + ac;

  const int bk = tid >> 3;              // 0..31 (l within k-tile)
  const int bc = (tid & 7) * 16;        // d col base
  const __bf16* gB = V + (size_t)n * LL * DD + n0 + bc;

  v8f acc[4][2];
#pragma unroll
  for (int i = 0; i < 4; ++i)
#pragma unroll
    for (int j = 0; j < 2; ++j)
#pragma unroll
      for (int v = 0; v < 8; ++v) acc[i][j][v] = 0.0f;

  for (int k0 = 0; k0 < LL; k0 += BK) {
    const float* srow = S + ((size_t)(n * LL + k0 + al)) * RR + m0 + ac;
    if (k0 + BK < LL) {
      __builtin_prefetch(srow + (size_t)BK * RR, 0, 1);
      __builtin_prefetch(gB + (size_t)(k0 + BK + bk) * DD, 0, 1);
    }
    // A: P_col^T tile 128(r) x 32(l): read S[l][r] coalesced over r, store transposed
#pragma unroll
    for (int u = 0; u < 16; u += 4) {
      v4f x  = *(const v4f*)(srow + u);
      v4f mm = *(const v4f*)(cmp + u);
      v4f ii = *(const v4f*)(cip + u);
#pragma unroll
      for (int q = 0; q < 4; ++q)
        sA[(ac + u + q) * LDT + al] = (__bf16)(__expf(x[q] - mm[q]) * ii[q]);
    }
    // B: lhs tile 32(l) x 128(d), stored transposed: sB[d][l]
    {
      const __bf16* pv = gB + (size_t)(k0 + bk) * DD;
      v8bf b0 = *(const v8bf*)pv;
      v8bf b1 = *(const v8bf*)(pv + 8);
#pragma unroll
      for (int u = 0; u < 8; ++u) sB[(bc + u) * LDT + bk] = b0[u];
#pragma unroll
      for (int u = 0; u < 8; ++u) sB[(bc + 8 + u) * LDT + bk] = b1[u];
    }
    __syncthreads();

    v16bf fa[4], fb[2];
#pragma unroll
    for (int i = 0; i < 4; ++i) fa[i] = load_frag_a(sA, wm + i * 16, lane);
#pragma unroll
    for (int j = 0; j < 2; ++j) fb[j] = load_frag_b(sB, wn + j * 16, lane);
#pragma unroll
    for (int i = 0; i < 4; ++i)
#pragma unroll
      for (int j = 0; j < 2; ++j)
        acc[i][j] = __builtin_amdgcn_wmma_f32_16x16x32_bf16(
            false, fa[i], false, fb[j], (short)0, acc[i][j], false, false);
    __syncthreads();
  }

  float* o = out + ((size_t)(n * RR + m0)) * (2 * DD) + DD + n0;
#pragma unroll
  for (int i = 0; i < 4; ++i)
#pragma unroll
    for (int j = 0; j < 2; ++j) {
      const int r0 = wm + i * 16 + ((lane >> 4) << 3);
      const int c  = wn + j * 16 + (lane & 15);
#pragma unroll
      for (int v = 0; v < 8; ++v)
        o[(size_t)(r0 + v) * (2 * DD) + c] = acc[i][j][v];
    }
}

// ---------------------------------------------------------------------------
// K5: pass-through halves of the concat outputs
// ---------------------------------------------------------------------------
__global__ __launch_bounds__(256) void bidir_copy_kernel(
    const float* __restrict__ lhs, const float* __restrict__ rhs, float* __restrict__ out)
{
  size_t i = (size_t)blockIdx.x * 256 + threadIdx.x;     // group of 4 floats
  const size_t half = (size_t)NB * LL * (DD / 4);
  v4f v;
  float* o;
  if (i < half) {
    v = ((const v4f*)lhs)[i];
    size_t e = i * 4;
    size_t row = e / DD; int d = (int)(e & (DD - 1));
    o = out + row * (2 * DD) + d;
  } else {
    size_t j = i - half;
    v = ((const v4f*)rhs)[j];
    size_t e = j * 4;
    size_t row = e / DD; int d = (int)(e & (DD - 1));
    o = out + (size_t)NB * LL * 2 * DD + row * (2 * DD) + d;
  }
  *(v4f*)o = v;
}

// ---------------------------------------------------------------------------
extern "C" void kernel_launch(void* const* d_in, const int* in_sizes, int n_in,
                              void* d_out, int out_size, void* d_ws, size_t ws_size,
                              hipStream_t stream) {
  (void)in_sizes; (void)n_in; (void)out_size; (void)ws_size;
  const float* lhs   = (const float*)d_in[0];
  const float* rhs   = (const float*)d_in[1];
  const float* w_lhs = (const float*)d_in[2];
  const float* w_rhs = (const float*)d_in[3];
  float* out = (float*)d_out;

  char* ws = (char*)d_ws;
  const size_t szT = (size_t)NB * LL * DD;                 // elems per bf16 tensor
  __bf16* lhsw = (__bf16*)ws;
  __bf16* rhsw = lhsw + szT;
  __bf16* lhsb = rhsw + szT;
  __bf16* rhsb = lhsb + szT;
  float*  S    = (float*)(rhsb + szT);
  float*  rmax = S + (size_t)NB * LL * RR;
  float*  rinv = rmax + (size_t)NB * LL;
  float*  cmax = rinv + (size_t)NB * LL;
  float*  cinv = cmax + (size_t)NB * RR;

  {
    unsigned groups = (unsigned)((size_t)NB * LL * DD / 4 / 256);
    bidir_scale_convert_kernel<<<dim3(groups), 256, 0, stream>>>(
        lhs, rhs, w_lhs, w_rhs, lhsw, rhsw, lhsb, rhsb);
  }
  bidir_scores_kernel<<<dim3(RR / BN, LL / BM, NB), 256, 0, stream>>>(lhsw, rhsw, S);
  bidir_row_stats_kernel<<<dim3(NB * LL), 256, 0, stream>>>(S, rmax, rinv);
  bidir_col_stats_kernel<<<dim3(RR / 256, NB), 256, 0, stream>>>(S, cmax, cinv);
  bidir_lhs_ctx_kernel<<<dim3(DD / BN, LL / BM, NB), 256, 0, stream>>>(S, rmax, rinv, rhsb, out);
  bidir_rhs_ctx_kernel<<<dim3(DD / BN, RR / BM, NB), 256, 0, stream>>>(
      S, cmax, cinv, lhsb, out + (size_t)NB * LL * 2 * DD);
  {
    unsigned groups = (unsigned)(2 * (size_t)NB * LL * DD / 4 / 256);
    bidir_copy_kernel<<<dim3(groups), 256, 0, stream>>>(lhs, rhs, out);
  }
}
